// TestQuantAttentionQuantPatternModel_27771258536801
// MI455X (gfx1250) — 8-bit matrix operand order repaired and checked by reading, compiled, not run on hardware
//
#include <hip/hip_runtime.h>
#include <math.h>

#define BATCH 64
#define HQ 32
#define HKV 8
#define DHEAD 128
#define HIDDEN 4096
#define KVBLOCK 16
#define BLOCKS_PER_SEQ 128
#define GQA 4
#define SM_SCALE 0.08838834764831845f

typedef __attribute__((ext_vector_type(16))) int   v16i_t;
typedef __attribute__((ext_vector_type(8)))  float v8f_t;

typedef int v4i_vec __attribute__((vector_size(16)));
typedef __attribute__((address_space(1))) v4i_vec glob_v4i;
typedef __attribute__((address_space(3))) v4i_vec lds_v4i;

#if __has_builtin(__builtin_amdgcn_global_load_async_to_lds_b128)
#define ATTN_ASYNC 1
#else
#define ATTN_ASYNC 0
#endif

__device__ __forceinline__ void wait_async_le(int n) {
#if ATTN_ASYNC
#if __has_builtin(__builtin_amdgcn_s_wait_asynccnt)
  if (n == 0) __builtin_amdgcn_s_wait_asynccnt(0);
  else        __builtin_amdgcn_s_wait_asynccnt(4);
#else
  if (n == 0) asm volatile("s_wait_asynccnt 0x0" ::: "memory");
  else        asm volatile("s_wait_asynccnt 0x4" ::: "memory");
#endif
#else
  (void)n;
#endif
}

__device__ __forceinline__ unsigned int f32_to_e4m3(float x) {
  float c = fminf(fmaxf(x, -448.f), 448.f);
  unsigned int u = __float_as_uint(c);
  unsigned int s = (u >> 24) & 0x80u;
  if ((u << 1) == 0u) return s;
  int e = (int)((u >> 23) & 0xFFu) - 127;
  unsigned int man = u & 0x7FFFFFu;
  if (e >= -6) {
    unsigned int m    = man >> 20;
    unsigned int rest = man & 0xFFFFFu;
    unsigned int rnd  = (rest > 0x80000u) || (rest == 0x80000u && (m & 1u));
    m += rnd;
    if (m == 8u) { m = 0u; e += 1; }
    if (e > 8) { e = 8; m = 6u; }
    return s | ((unsigned int)(e + 7) << 3) | m;
  }
  if (e < -13) return s;
  int sh = -6 - e;
  unsigned int full = 0x800000u | man;
  unsigned int m    = full >> (20 + sh);
  unsigned int hlf  = 1u << (19 + sh);
  unsigned int rem  = full & ((1u << (20 + sh)) - 1u);
  unsigned int rnd  = (rem > hlf) || (rem == hlf && (m & 1u));
  m += rnd;
  if (m >= 8u) return s | (1u << 3);
  return s | m;
}

__global__ void __launch_bounds__(256) quant_w_kernel(
    const float* __restrict__ W, unsigned char* __restrict__ Wq) {
  int i = blockIdx.x * blockDim.x + threadIdx.x;
  float4 wv = ((const float4*)W)[i];
  unsigned int packed = f32_to_e4m3(wv.x)
                      | (f32_to_e4m3(wv.y) << 8)
                      | (f32_to_e4m3(wv.z) << 16)
                      | (f32_to_e4m3(wv.w) << 24);
  ((unsigned int*)Wq)[i] = packed;
}

__device__ __forceinline__ void stage_chunk(
    int c, int slot, int tid, int b, int h, int S,
    const float* __restrict__ knew, const float* __restrict__ vnew,
    const float* __restrict__ kv_cache, const int* __restrict__ block_table,
    float (*lds)[2][KVBLOCK][DHEAD]) {
  int blk = block_table[b * BLOCKS_PER_SEQ + c];
#pragma unroll
  for (int t = 0; t < 2; ++t) {
#pragma unroll
    for (int j = 0; j < 2; ++j) {
      int e   = tid + j * 256;
      int row = e >> 5;
      int col = e & 31;
      int p   = c * KVBLOCK + row;
      const float* src;
      if (p == S) {
        src = (t ? vnew : knew) + (size_t)b * (HKV * DHEAD) + h * DHEAD + col * 4;
      } else {
        src = kv_cache + (((size_t)blk * 2 + t) * KVBLOCK + row) * (HKV * DHEAD)
              + h * DHEAD + col * 4;
      }
      float* dst = &lds[slot][t][row][col * 4];
#if ATTN_ASYNC
      __builtin_amdgcn_global_load_async_to_lds_b128(
          (glob_v4i*)src, (lds_v4i*)dst, 0, 0);
#else
      *(float4*)dst = *(const float4*)src;
#endif
    }
  }
}

__global__ void __launch_bounds__(256) attn_decode_kernel(
    const float* __restrict__ q, const float* __restrict__ knew,
    const float* __restrict__ vnew, const float* __restrict__ kv_cache,
    const int* __restrict__ block_table, const int* __restrict__ seq_lens,
    unsigned char* __restrict__ attn_q8) {
  __shared__ __align__(16) float lds_kv[2][2][KVBLOCK][DHEAD];
  __shared__ float s_m[8];
  __shared__ float s_l[8];
  __shared__ __align__(16) float s_acc[8][GQA][DHEAD];

  const int b    = blockIdx.x;
  const int h    = blockIdx.y;
  const int tid  = threadIdx.x;
  const int lane = tid & 31;
  const int wave = tid >> 5;

  const int S       = seq_lens[b];
  const int npos    = S + 1;
  const int nchunks = (npos + KVBLOCK - 1) >> 4;

  float4 qr[GQA];
#pragma unroll
  for (int r = 0; r < GQA; ++r)
    qr[r] = *(const float4*)(q + (size_t)b * HIDDEN + h * (GQA * DHEAD)
                               + r * DHEAD + lane * 4);

  float m = -1e30f, l = 0.f;
  float4 acc[GQA];
#pragma unroll
  for (int r = 0; r < GQA; ++r) acc[r] = make_float4(0.f, 0.f, 0.f, 0.f);

  stage_chunk(0, 0, tid, b, h, S, knew, vnew, kv_cache, block_table, lds_kv);

  for (int c = 0; c < nchunks; ++c) {
    int slot = c & 1;
    if (c + 1 < nchunks) {
      stage_chunk(c + 1, slot ^ 1, tid, b, h, S, knew, vnew, kv_cache,
                  block_table, lds_kv);
      wait_async_le(4);
    } else {
      wait_async_le(0);
    }
    __syncthreads();

#pragma unroll
    for (int j = 0; j < 2; ++j) {
      int row = wave * 2 + j;
      int p   = c * KVBLOCK + row;
      if (p < npos) {
        float4 k4 = *(const float4*)&lds_kv[slot][0][row][lane * 4];
        float sc[GQA];
#pragma unroll
        for (int r = 0; r < GQA; ++r) {
          float d0 = qr[r].x * k4.x + qr[r].y * k4.y
                   + qr[r].z * k4.z + qr[r].w * k4.w;
#pragma unroll
          for (int o = 16; o > 0; o >>= 1) d0 += __shfl_xor(d0, o, 32);
          sc[r] = d0 * SM_SCALE;
        }
        float mn = m;
#pragma unroll
        for (int r = 0; r < GQA; ++r) mn = fmaxf(mn, sc[r]);
        float corr = __expf(m - mn);
        float wgt[GQA];
        float wsum = 0.f;
#pragma unroll
        for (int r = 0; r < GQA; ++r) { wgt[r] = __expf(sc[r] - mn); wsum += wgt[r]; }
        l = l * corr + wsum;
        float4 v4 = *(const float4*)&lds_kv[slot][1][row][lane * 4];
#pragma unroll
        for (int r = 0; r < GQA; ++r) {
          acc[r].x = acc[r].x * corr + wgt[r] * v4.x;
          acc[r].y = acc[r].y * corr + wgt[r] * v4.y;
          acc[r].z = acc[r].z * corr + wgt[r] * v4.z;
          acc[r].w = acc[r].w * corr + wgt[r] * v4.w;
        }
        m = mn;
      }
    }
    __syncthreads();
  }

  if (lane == 0) { s_m[wave] = m; s_l[wave] = l; }
#pragma unroll
  for (int r = 0; r < GQA; ++r)
    ((float4*)&s_acc[wave][r][0])[lane] = acc[r];
  __syncthreads();

#pragma unroll
  for (int t = 0; t < 2; ++t) {
    int i = tid * 2 + t;
    int r = i >> 7, d = i & 127;
    float M = -1e30f;
#pragma unroll
    for (int w = 0; w < 8; ++w) M = fmaxf(M, s_m[w]);
    float L = 0.f, o = 0.f;
#pragma unroll
    for (int w = 0; w < 8; ++w) {
      float e = __expf(s_m[w] - M);
      L += s_l[w] * e;
      o += s_acc[w][r][d] * e;
    }
    o = o / L;
    attn_q8[(size_t)b * HIDDEN + h * (GQA * DHEAD) + i] =
        (unsigned char)f32_to_e4m3(o);
  }
}

__global__ void __launch_bounds__(128) gemm_fp8_wmma_kernel(
    const unsigned char* __restrict__ Aq, const unsigned char* __restrict__ Wq,
    float* __restrict__ out) {
  const int lane = threadIdx.x & 31;
  const int wv   = threadIdx.x >> 5;
  const int m0   = blockIdx.x * 16;
  const int n0   = (blockIdx.y * 4 + wv) * 16;
  const int l16  = lane & 15;
  const int hf   = lane >> 4;

  const unsigned char* arow = Aq + (size_t)(m0 + l16) * HIDDEN;
  const unsigned char* brow = Wq + (size_t)(n0 + l16) * HIDDEN;

  v8f_t c = {};
  for (int k0 = 0; k0 < HIDDEN; k0 += 128) {
    __builtin_prefetch(arow + k0 + 256, 0, 1);
    __builtin_prefetch(brow + k0 + 256, 0, 1);
    v16i_t av, bv;
#pragma unroll
    for (int r = 0; r < 16; ++r) {
      int ka = ((r >> 3) << 6) + (((r >> 1) & 3) << 4) + ((r & 1) << 2) + (hf << 3);
      av[r] = *(const int*)(arow + k0 + ka);
      int kb = ((r >> 3) << 6) + (((r >> 1) & 3) << 4) + ((r & 1) << 2) + (hf << 3);
      bv[r] = *(const int*)(brow + k0 + kb);
    }
    c = __builtin_amdgcn_wmma_f32_16x16x128_fp8_fp8(av, bv, (short)0, c,
                                                    false, false);
  }
#pragma unroll
  for (int r = 0; r < 8; ++r)
    out[(size_t)(m0 + hf * 8 + r) * HIDDEN + n0 + l16] = c[r];
}

extern "C" void kernel_launch(void* const* d_in, const int* in_sizes, int n_in,
                              void* d_out, int out_size, void* d_ws,
                              size_t ws_size, hipStream_t stream) {
  (void)in_sizes; (void)n_in; (void)out_size; (void)ws_size;
  const float* q         = (const float*)d_in[0];
  const float* k         = (const float*)d_in[1];
  const float* v         = (const float*)d_in[2];
  const float* kv_cache  = (const float*)d_in[3];
  const float* W         = (const float*)d_in[4];
  const int*   block_tab = (const int*)d_in[5];
  const int*   seq_lens  = (const int*)d_in[6];
  float*       out       = (float*)d_out;

  unsigned char* Wq = (unsigned char*)d_ws;
  unsigned char* Aq = Wq + (size_t)HIDDEN * HIDDEN;

  quant_w_kernel<<<dim3((HIDDEN * HIDDEN) / (256 * 4)), dim3(256), 0, stream>>>(W, Wq);
  attn_decode_kernel<<<dim3(BATCH, HKV), dim3(256), 0, stream>>>(
      q, k, v, kv_cache, block_tab, seq_lens, Aq);
  gemm_fp8_wmma_kernel<<<dim3(BATCH / 16, HIDDEN / (16 * 4)), dim3(128), 0, stream>>>(
      Aq, Wq, out);
}
